// DissipativeSimplestRINN_25348896981725
// MI455X (gfx1250) — compile-verified, matrix operand order checked by reading, not run on hardware
//
#include <hip/hip_runtime.h>

typedef _Float16 v16h __attribute__((ext_vector_type(16)));
typedef _Float16 v8h  __attribute__((ext_vector_type(8)));
typedef _Float16 v4h  __attribute__((ext_vector_type(4)));
typedef float    v8f  __attribute__((ext_vector_type(8)));

#define B_TOTAL   2048
#define T_STEPS   64
#define INP_D     32
#define STATE_D   64
#define NONLIN_D  128
#define OUT_PAIR  16      // u(8) concat log_stds(8)
#define N_FP      30
#define DT_C      0.01f

#define WAVES_PB  4                 // 4 waves cooperate on one 16-row batch tile
#define BLOCK_T   (WAVES_PB * 32)   // 128 threads
#define GRID_B    (B_TOTAL / 16)    // 128 blocks

// ---- LDS strides in halves (multiples of 8 halves = 16B; dword row strides
// 68/52/116/180 mod 64 give 16 distinct banks for 16 consecutive rows) ----
#define S_DVW 136   // Dvw  [N=128][K=128 pad]
#define S_VC  104   // [Cv;Dvy] [N=128][K=96 pad]
#define S_G   232   // [A,Bw,By | Cu,Duw,Duy] [N=80][K=224 pad]
#define S_XYW 360   // staging row: x(64) | wbuf0(128) | wbuf1(128) | y(32) pad->360

#define XOFF  0
#define W0OFF 64
#define W1OFF 192
#define YOFF  320

#define OFF_DVW 0
#define OFF_VC  (OFF_DVW + 128 * S_DVW)
#define OFF_G   (OFF_VC  + 128 * S_VC)
#define OFF_XYW (OFF_G   + 80  * S_G)
#define SMEM_HALVES (OFF_XYW + 16 * S_XYW)
#define SMEM_BYTES  (SMEM_HALVES * 2)          // ~108 KB < 320 KB/WGP

static __device__ __forceinline__ v8f wmma16(v16h a, v16h b, v8f c) {
  // D = A(16x32 f16) * B(32x16 f16) + C(16x16 f32)
  return __builtin_amdgcn_wmma_f32_16x16x32_f16(
      /*neg_a=*/false, a, /*neg_b=*/false, b,
      /*c_mod=*/(short)0, c, /*reuse_a=*/false, /*reuse_b=*/false);
}

// Build a 16-bit A/B WMMA operand from LDS; p = row_ptr + kbase + laneHi*8.
// halves 0..7 = K(kbase+hi*8..+7), halves 8..15 = K(kbase+16+hi*8..+7).
static __device__ __forceinline__ v16h ld_op(const _Float16* p) {
  v8h lo = *(const v8h*)(p);
  v8h hi = *(const v8h*)(p + 16);
  v16h r;
#pragma unroll
  for (int i = 0; i < 8; ++i) { r[i] = lo[i]; r[i + 8] = hi[i]; }
  return r;
}

static __device__ __forceinline__ float dtanh(float x) {
#if defined(__gfx1250__) && __has_builtin(__builtin_amdgcn_tanhf)
  return __builtin_amdgcn_tanhf(x);          // v_tanh_f32
#else
  float e = __builtin_amdgcn_exp2f(2.8853900817779268f * x);  // exp(2x)
  return 1.0f - 2.0f / (e + 1.0f);
#endif
}

__global__ __launch_bounds__(BLOCK_T, 1)
void rinn_scan_kernel(const float* __restrict__ obs,   const float* __restrict__ x0,
                      const float* __restrict__ A_T,   const float* __restrict__ Bw_T,
                      const float* __restrict__ By_T,  const float* __restrict__ Cv_T,
                      const float* __restrict__ Dvw_T, const float* __restrict__ Dvy_T,
                      const float* __restrict__ Cu_T,  const float* __restrict__ Duw_T,
                      const float* __restrict__ Duy_T, const float* __restrict__ lstd_g,
                      float* __restrict__ out) {
  extern __shared__ _Float16 smem[];
  _Float16* s_dvw = smem + OFF_DVW;
  _Float16* s_vc  = smem + OFF_VC;
  _Float16* s_g   = smem + OFF_G;
  _Float16* xyw   = smem + OFF_XYW;   // one 16-row staging tile, shared by 4 waves

  const int tid = threadIdx.x;

  // ---- one-time weight load: f32 global -> f16 LDS in [N][K] layout ----
  for (int idx = tid; idx < 128 * 128; idx += BLOCK_T) {
    int n = idx >> 7, k = idx & 127;
    s_dvw[n * S_DVW + k] = (_Float16)Dvw_T[k * NONLIN_D + n];
  }
  for (int idx = tid; idx < 128 * 96; idx += BLOCK_T) {
    int n = idx / 96, k = idx % 96;
    float v = (k < 64) ? Cv_T[k * NONLIN_D + n] : Dvy_T[(k - 64) * NONLIN_D + n];
    s_vc[n * S_VC + k] = (_Float16)v;
  }
  for (int idx = tid; idx < 80 * 224; idx += BLOCK_T) {
    int n = idx / 224, k = idx % 224;
    float v;
    if (n < 64) {
      v = (k < 64)  ? A_T[k * STATE_D + n]
        : (k < 192) ? Bw_T[(k - 64) * STATE_D + n]
                    : By_T[(k - 192) * STATE_D + n];
    } else if (n < 72) {
      int m = n - 64;
      v = (k < 64)  ? Cu_T[k * 8 + m]
        : (k < 192) ? Duw_T[(k - 64) * 8 + m]
                    : Duy_T[(k - 192) * 8 + m];
    } else {
      v = 0.0f;
    }
    s_g[n * S_G + k] = (_Float16)v;
  }
  __syncthreads();

  const int lane   = tid & 31;
  const int wave   = tid >> 5;      // 0..3 : owns N-tiles {2w, 2w+1} of NONLIN,
                                    //        state tile w, (wave0: u tile)
  const int laneM  = lane & 15;
  const int laneHi = lane >> 4;
  const int rowBase = blockIdx.x * 16;

  const _Float16* myArow = xyw + laneM * S_XYW;

  // ---- loop-invariant Dvw B-tiles, register resident: 8 tiles = 64 VGPRs ----
  v16h bdvw[2][4];
#pragma unroll
  for (int j = 0; j < 2; ++j)
#pragma unroll
    for (int kt = 0; kt < 4; ++kt)
      bdvw[j][kt] =
          ld_op(s_dvw + ((2 * wave + j) * 16 + laneM) * S_DVW + kt * 32 + laneHi * 8);

  const float stdv = lstd_g[laneM & 7];

  // state tile (cols 16*wave .. +15) in fp32 C-layout: elem(M=i+8*laneHi, col)
  v8f xs;
#pragma unroll
  for (int i = 0; i < 8; ++i)
    xs[i] = x0[(size_t)(rowBase + i + 8 * laneHi) * STATE_D + wave * 16 + laneM];

#pragma unroll 1
  for (int t = 0; t < T_STEPS; ++t) {
    __syncthreads();   // prior timestep's out-GEMM reads of x/y staging are done

    // stage my x tile (f16 scatter, disjoint per lane/wave)
#pragma unroll
    for (int i = 0; i < 8; ++i)
      xyw[(i + 8 * laneHi) * S_XYW + XOFF + wave * 16 + laneM] = (_Float16)xs[i];

    // stage my obs slice: cols [8*wave, 8*wave+8), 4 halves per lane
    {
      const float* op =
          obs + ((size_t)(rowBase + laneM) * T_STEPS + t) * INP_D + wave * 8 + laneHi * 4;
      v4h h;
#pragma unroll
      for (int c = 0; c < 4; ++c) h[c] = (_Float16)op[c];
      *(v4h*)(xyw + laneM * S_XYW + YOFF + wave * 8 + laneHi * 4) = h;
    }
    __syncthreads();   // staging visible to all 4 waves

    // ---- v_const = [x|y] @ [Cv;Dvy] for my 2 N-tiles ----
    v8f vca[2];
    vca[0] = (v8f)(0.0f);
    vca[1] = (v8f)(0.0f);
    {
      const int kA[3] = {XOFF, XOFF + 32, YOFF};
      const int kB[3] = {0, 32, 64};
#pragma unroll
      for (int kk = 0; kk < 3; ++kk) {
        v16h a = ld_op(myArow + kA[kk] + laneHi * 8);
#pragma unroll
        for (int j = 0; j < 2; ++j) {
          v16h b = ld_op(s_vc + ((2 * wave + j) * 16 + laneM) * S_VC + kB[kk] + laneHi * 8);
          vca[j] = wmma16(a, b, vca[j]);
        }
      }
    }

    // ---- fixed point, iteration 0 (w0 = 0): w = tanh(v_const) -> wbuf0 ----
#pragma unroll
    for (int j = 0; j < 2; ++j)
#pragma unroll
      for (int i = 0; i < 8; ++i)
        xyw[(i + 8 * laneHi) * S_XYW + W0OFF + (2 * wave + j) * 16 + laneM] =
            (_Float16)dtanh(vca[j][i]);

    // ---- iterations 1..29: ping-pong LDS w buffers, one barrier each ----
#pragma unroll 1
    for (int it = 1; it < N_FP; ++it) {
      __syncthreads();                       // all stores to read-buffer visible
      const int roff = (it & 1) ? W0OFF : W1OFF;
      const int woff = (it & 1) ? W1OFF : W0OFF;

      v16h at[4];
#pragma unroll
      for (int kt = 0; kt < 4; ++kt)
        at[kt] = ld_op(myArow + roff + kt * 32 + laneHi * 8);

      v8f wacc[2];
#pragma unroll
      for (int j = 0; j < 2; ++j) wacc[j] = wmma16(at[0], bdvw[j][0], vca[j]);
#pragma unroll
      for (int kt = 1; kt < 4; ++kt)
#pragma unroll
        for (int j = 0; j < 2; ++j) wacc[j] = wmma16(at[kt], bdvw[j][kt], wacc[j]);

#pragma unroll
      for (int j = 0; j < 2; ++j)
#pragma unroll
        for (int i = 0; i < 8; ++i)
          xyw[(i + 8 * laneHi) * S_XYW + woff + (2 * wave + j) * 16 + laneM] =
              (_Float16)dtanh(wacc[j][i]);
    }
    __syncthreads();   // final w (in wbuf1, since N_FP-1=29 is odd) visible

    // ---- [x|w|y] @ G : my state tile (+ wave0: u tile) ----
    {
      const int kAo[7] = {XOFF, XOFF + 32, W1OFF, W1OFF + 32, W1OFF + 64, W1OFF + 96, YOFF};
      v8f ox = (v8f)(0.0f);
      v8f ou = (v8f)(0.0f);
#pragma unroll
      for (int kt = 0; kt < 7; ++kt) {
        v16h a = ld_op(myArow + kAo[kt] + laneHi * 8);
        v16h b = ld_op(s_g + (wave * 16 + laneM) * S_G + kt * 32 + laneHi * 8);
        ox = wmma16(a, b, ox);
        if (wave == 0) {   // uniform per wave: EXEC stays all-ones inside
          v16h bu = ld_op(s_g + (64 + laneM) * S_G + kt * 32 + laneHi * 8);
          ou = wmma16(a, bu, ou);
        }
      }
#pragma unroll
      for (int i = 0; i < 8; ++i) xs[i] += DT_C * ox[i];

      if (wave == 0) {
#pragma unroll
        for (int i = 0; i < 8; ++i) {
          int m = i + 8 * laneHi;
          float val = (laneM < 8) ? ou[i] : stdv;
          out[((size_t)(rowBase + m) * T_STEPS + t) * OUT_PAIR + laneM] = val;
        }
      }
    }
  }

  // x_final after out[B,T,16]
  const size_t xf_off = (size_t)B_TOTAL * T_STEPS * OUT_PAIR;
#pragma unroll
  for (int i = 0; i < 8; ++i)
    out[xf_off + (size_t)(rowBase + i + 8 * laneHi) * STATE_D + wave * 16 + laneM] = xs[i];
}

extern "C" void kernel_launch(void* const* d_in, const int* in_sizes, int n_in,
                              void* d_out, int out_size, void* d_ws, size_t ws_size,
                              hipStream_t stream) {
  (void)in_sizes; (void)n_in; (void)d_ws; (void)ws_size; (void)out_size;
  const float* obs   = (const float*)d_in[0];
  const float* x0    = (const float*)d_in[1];
  const float* A_T   = (const float*)d_in[2];
  const float* Bw_T  = (const float*)d_in[3];
  const float* By_T  = (const float*)d_in[4];
  const float* Cv_T  = (const float*)d_in[5];
  const float* Dvw_T = (const float*)d_in[6];
  const float* Dvy_T = (const float*)d_in[7];
  const float* Cu_T  = (const float*)d_in[8];
  const float* Duw_T = (const float*)d_in[9];
  const float* Duy_T = (const float*)d_in[10];
  const float* lstd  = (const float*)d_in[11];
  float* out = (float*)d_out;

  rinn_scan_kernel<<<GRID_B, BLOCK_T, SMEM_BYTES, stream>>>(
      obs, x0, A_T, Bw_T, By_T, Cv_T, Dvw_T, Dvy_T, Cu_T, Duw_T, Duy_T, lstd, out);
}
